// DecoderRNN_38628935860292
// MI455X (gfx1250) — matrix operand order repaired and checked by reading, not compiled, not run on hardware
//
#include <hip/hip_runtime.h>

#define B_ 128
#define S_ 256
#define H_ 512
#define V_ 50000
#define E_ 256

typedef __attribute__((ext_vector_type(16))) __bf16 vbf16;
typedef __attribute__((ext_vector_type(8)))  __bf16 vbf8;
typedef __attribute__((ext_vector_type(4)))  __bf16 vbf4;
typedef __attribute__((ext_vector_type(8)))  float  v8f;

__device__ __forceinline__ vbf16 ld_frag(const __bf16* p0, const __bf16* p1) {
  vbf8 a = *(const vbf8*)p0;
  vbf8 b = *(const vbf8*)p1;
  vbf16 r;
#pragma unroll
  for (int i = 0; i < 8; ++i) { r[i] = a[i]; r[i + 8] = b[i]; }
  return r;
}

// C[M,N] = A[M,K](fp32,row-major,lda) @ Bw^T + bias
// Bw[n,k] at Bw[n*brs + k*bcs].  BVEC=true: bcs==1, vectorized b128 staging.
// fp32 -> bf16 during staging; WMMA bf16, fp32 accumulate.
// Double-buffered LDS pipeline: prefetch next K-tile into registers while
// WMMAs consume the current buffer; one barrier per K-step.
// Staging is branchless: OOB rows are clamped (their products only land in
// C rows/cols masked off at the guarded store).
template <bool BVEC>
__global__ __launch_bounds__(128) void k_gemm_bf16wmma(
    const float* __restrict__ A, int lda,
    const float* __restrict__ Bw, long brs, int bcs,
    const float* __restrict__ bias,
    float* __restrict__ C, int ldc,
    int M, int N, int K)
{
  constexpr int AS = 40;                 // padded LDS row stride (bf16 units)
  __shared__ __bf16 la[2][64 * AS];
  __shared__ __bf16 lb[2][64 * AS];
  const int tid  = threadIdx.x;
  const int lane = tid & 31;
  const int wv   = tid >> 5;             // wave 0..3 (wave32)
  const int tm   = blockIdx.y * 64;
  const int tn   = blockIdx.x * 64;

  int rr[4], cc[4];                      // per-thread staging coords (4x float4)
#pragma unroll
  for (int j = 0; j < 4; ++j) {
    int idx4 = j * 128 + tid;
    rr[j] = idx4 >> 3;
    cc[j] = (idx4 & 7) << 2;
  }

  float4 ra[4], rb[4];
  float rbs[16];

  auto load_regs = [&](int k0) {
#pragma unroll
    for (int j = 0; j < 4; ++j) {
      int gr = min(tm + rr[j], M - 1);
      ra[j] = *(const float4*)(A + (size_t)gr * lda + k0 + cc[j]);
    }
    if (BVEC) {
#pragma unroll
      for (int j = 0; j < 4; ++j) {
        int gn = min(tn + rr[j], N - 1);
        rb[j] = *(const float4*)(Bw + (size_t)gn * brs + k0 + cc[j]);
      }
    } else {
#pragma unroll
      for (int i = 0; i < 16; ++i) {
        int idx = i * 128 + tid;
        int r = idx >> 5, c = idx & 31;
        int gn = min(tn + r, N - 1);
        rbs[i] = Bw[(size_t)gn * brs + (size_t)(k0 + c) * bcs];
      }
    }
  };

  auto store_lds = [&](int buf) {
#pragma unroll
    for (int j = 0; j < 4; ++j) {
      vbf4 p = { (__bf16)ra[j].x, (__bf16)ra[j].y, (__bf16)ra[j].z, (__bf16)ra[j].w };
      *(vbf4*)(&la[buf][rr[j] * AS + cc[j]]) = p;
    }
    if (BVEC) {
#pragma unroll
      for (int j = 0; j < 4; ++j) {
        vbf4 p = { (__bf16)rb[j].x, (__bf16)rb[j].y, (__bf16)rb[j].z, (__bf16)rb[j].w };
        *(vbf4*)(&lb[buf][rr[j] * AS + cc[j]]) = p;
      }
    } else {
#pragma unroll
      for (int i = 0; i < 16; ++i) {
        int idx = i * 128 + tid;
        int r = idx >> 5, c = idx & 31;
        lb[buf][r * AS + c] = (__bf16)rbs[i];
      }
    }
  };

  v8f acc[4];
#pragma unroll
  for (int cb = 0; cb < 4; ++cb)
#pragma unroll
    for (int r = 0; r < 8; ++r) acc[cb][r] = 0.f;

  load_regs(0);
  store_lds(0);

  const int mrow = wv * 16 + (lane & 15);
  const int ka = (lane >= 16) ? 16 : 0;  // A frag: 16 contiguous K at ka
  const int kb = (lane >= 16) ? 16 : 0;  // B frag: 16 contiguous K at kb

  int buf = 0;
  for (int k0 = 0; k0 < K; k0 += 32, buf ^= 1) {
    __syncthreads();
    const bool more = (k0 + 32) < K;
    if (more) load_regs(k0 + 32);        // next tile in flight during WMMAs
    vbf16 af = ld_frag(&la[buf][mrow * AS + ka], &la[buf][mrow * AS + ka + 8]);
#pragma unroll
    for (int cb = 0; cb < 4; ++cb) {
      const int nrow = cb * 16 + (lane & 15);
      vbf16 bf = ld_frag(&lb[buf][nrow * AS + kb], &lb[buf][nrow * AS + kb + 8]);
      acc[cb] = __builtin_amdgcn_wmma_f32_16x16x32_bf16(
          false, af, false, bf, (short)0, acc[cb], false, false);
    }
    if (more) store_lds(buf ^ 1);
  }

  // C/D layout: VGPR r -> M = r + (lane>=16 ? 8 : 0), N = lane%16
  const int col_l = lane & 15;
  const int row_off = (lane >> 4) * 8;
#pragma unroll
  for (int cb = 0; cb < 4; ++cb) {
    int gc = tn + cb * 16 + col_l;
    if (gc >= N) continue;
    float bv = bias[gc];
#pragma unroll
    for (int r = 0; r < 8; ++r) {
      int gr = tm + wv * 16 + row_off + r;
      if (gr < M) C[(size_t)gr * ldc + gc] = acc[cb][r] + bv;
    }
  }
}

// w_raw[b,s] = sum_h tanh(enc_feat[b,s,h] + dec_feat[b,h] + cov_feat[b,s]) * v[b,h]
__global__ __launch_bounds__(256) void k_attn_score(
    const float* __restrict__ enc_feat, const float* __restrict__ dec_feat,
    const float* __restrict__ cov_feat, const float* __restrict__ v,
    const int* __restrict__ mask, float* __restrict__ w_raw)
{
  __shared__ float sdec[H_];
  __shared__ float sv[H_];
  const int b = blockIdx.x, tid = threadIdx.x;
  for (int h = tid; h < H_; h += 256) {
    sdec[h] = dec_feat[b * H_ + h];
    sv[h]   = v[b * H_ + h];
  }
  __syncthreads();
  const int lane = tid & 31, wv = tid >> 5;
  const float4* sd4 = (const float4*)sdec;
  const float4* sv4 = (const float4*)sv;
  for (int s = wv; s < S_; s += 8) {
    const float cf = cov_feat[b * S_ + s];
    const float4* row4 = (const float4*)(enc_feat + (size_t)(b * S_ + s) * H_);
    float acc = 0.f;
#pragma unroll
    for (int i = 0; i < H_ / 4 / 32; ++i) {       // 4 iterations of float4
      int p = i * 32 + lane;
      float4 e = row4[p], d = sd4[p], vv = sv4[p];
      acc += tanhf(e.x + d.x + cf) * vv.x;
      acc += tanhf(e.y + d.y + cf) * vv.y;
      acc += tanhf(e.z + d.z + cf) * vv.z;
      acc += tanhf(e.w + d.w + cf) * vv.w;
    }
#pragma unroll
    for (int off = 16; off > 0; off >>= 1)
      acc += __shfl_xor(acc, off, 32);
    if (lane == 0)
      w_raw[b * S_ + s] = (mask[b * S_ + s] == 1) ? acc : -1e30f;
  }
}

__global__ __launch_bounds__(256) void k_softmax_cov(
    const float* __restrict__ w_raw, const float* __restrict__ cov_in,
    float* __restrict__ w_out, float* __restrict__ cov_out,
    float* __restrict__ w_aux)
{
  __shared__ float red[256];
  const int b = blockIdx.x, s = threadIdx.x;
  float x = w_raw[b * S_ + s];
  red[s] = x; __syncthreads();
  for (int off = 128; off > 0; off >>= 1) {
    if (s < off) red[s] = fmaxf(red[s], red[s + off]);
    __syncthreads();
  }
  float m = red[0]; __syncthreads();
  float e = expf(x - m);
  red[s] = e; __syncthreads();
  for (int off = 128; off > 0; off >>= 1) {
    if (s < off) red[s] += red[s + off];
    __syncthreads();
  }
  float wv = e / red[0];
  w_out[b * S_ + s] = wv;
  cov_out[b * S_ + s] = cov_in[b * S_ + s] + wv;
  if (w_aux) w_aux[b * S_ + s] = wv;
}

// ctx[b,h] = sum_s w[b,s] * enc[b,s,h]; float2 per thread, strided output rows
__global__ __launch_bounds__(256) void k_ctx(
    const float* __restrict__ w, const float* __restrict__ enc,
    float* __restrict__ out, int ostride, float* __restrict__ aux)
{
  __shared__ float sw[S_];
  const int b = blockIdx.x, tid = threadIdx.x;
  sw[tid] = w[b * S_ + tid];
  __syncthreads();
  float ax = 0.f, ay = 0.f;
  const float* eb = enc + (size_t)b * S_ * H_ + 2 * tid;
  for (int s = 0; s < S_; ++s) {
    float ww = sw[s];
    const float2 e = *(const float2*)(eb + (size_t)s * H_);
    ax += ww * e.x;
    ay += ww * e.y;
  }
  float2 r; r.x = ax; r.y = ay;
  *(float2*)(out + (size_t)b * ostride + 2 * tid) = r;
  if (aux) *(float2*)(aux + (size_t)b * H_ + 2 * tid) = r;
}

__global__ void k_embed(const int* __restrict__ token,
                        const float* __restrict__ emb,
                        float* __restrict__ xcat)
{
  int b = blockIdx.x, e = threadIdx.x;
  xcat[(size_t)b * (E_ + H_) + e] = emb[(size_t)token[b] * E_ + e];
}

__global__ __launch_bounds__(256) void k_gru(
    const float* __restrict__ gi, const float* __restrict__ gh,
    const float* __restrict__ hprev,
    float* __restrict__ h_new, float* __restrict__ h_out, float* __restrict__ cat2)
{
  int idx = blockIdx.x * 256 + threadIdx.x;
  if (idx >= B_ * H_) return;
  int b = idx / H_, h = idx % H_;
  float ir = gi[b * 3 * H_ + h], iz = gi[b * 3 * H_ + H_ + h], in_ = gi[b * 3 * H_ + 2 * H_ + h];
  float hr = gh[b * 3 * H_ + h], hz = gh[b * 3 * H_ + H_ + h], hn = gh[b * 3 * H_ + 2 * H_ + h];
  float r = 1.f / (1.f + expf(-(ir + hr)));
  float z = 1.f / (1.f + expf(-(iz + hz)));
  float ng = tanhf(in_ + r * hn);
  float hp = hprev[idx];
  float hv = (1.f - z) * ng + z * hp;
  h_new[idx] = hv;
  h_out[idx] = hv;
  cat2[(size_t)b * (2 * H_) + h] = hv;
}

__global__ __launch_bounds__(256) void k_logsoftmax(float* __restrict__ logits)
{
  __shared__ float red[256];
  const int b = blockIdx.x, tid = threadIdx.x;
  float4* row4 = (float4*)(logits + (size_t)b * V_);
  const int n4 = V_ / 4;                           // 12500, V_ % 4 == 0
  float m = -3.4e38f;
  for (int i = tid; i < n4; i += 256) {
    float4 x = row4[i];
    m = fmaxf(m, fmaxf(fmaxf(x.x, x.y), fmaxf(x.z, x.w)));
  }
  red[tid] = m; __syncthreads();
  for (int off = 128; off > 0; off >>= 1) {
    if (tid < off) red[tid] = fmaxf(red[tid], red[tid + off]);
    __syncthreads();
  }
  m = red[0]; __syncthreads();
  float sum = 0.f;
  for (int i = tid; i < n4; i += 256) {
    float4 x = row4[i];
    sum += expf(x.x - m) + expf(x.y - m) + expf(x.z - m) + expf(x.w - m);
  }
  red[tid] = sum; __syncthreads();
  for (int off = 128; off > 0; off >>= 1) {
    if (tid < off) red[tid] += red[tid + off];
    __syncthreads();
  }
  float ls = logf(red[0]) + m;
  for (int i = tid; i < n4; i += 256) {
    float4 x = row4[i];
    x.x -= ls; x.y -= ls; x.z -= ls; x.w -= ls;
    row4[i] = x;
  }
}

extern "C" void kernel_launch(void* const* d_in, const int* in_sizes, int n_in,
                              void* d_out, int out_size, void* d_ws, size_t ws_size,
                              hipStream_t stream)
{
  (void)in_sizes; (void)n_in; (void)out_size; (void)ws_size;
  const int*   token     = (const int*)d_in[0];
  const int*   mask      = (const int*)d_in[1];
  const float* enc       = (const float*)d_in[2];
  const float* hidden    = (const float*)d_in[3];
  const float* coverage  = (const float*)d_in[4];
  const float* attnLin_w = (const float*)d_in[5];
  const float* attnLin_b = (const float*)d_in[6];
  const float* decAttn_w = (const float*)d_in[7];
  const float* decAttn_b = (const float*)d_in[8];
  const float* cvg_w     = (const float*)d_in[9];
  const float* cvg_b     = (const float*)d_in[10];
  const float* vvec      = (const float*)d_in[11];
  const float* emb       = (const float*)d_in[12];
  const float* newIn_w   = (const float*)d_in[13];
  const float* newIn_b   = (const float*)d_in[14];
  const float* gru_w_ih  = (const float*)d_in[15];
  const float* gru_w_hh  = (const float*)d_in[16];
  const float* gru_b_ih  = (const float*)d_in[17];
  const float* gru_b_hh  = (const float*)d_in[18];
  const float* preOut_w  = (const float*)d_in[19];
  const float* preOut_b  = (const float*)d_in[20];
  const float* out_w     = (const float*)d_in[21];
  const float* out_b     = (const float*)d_in[22];

  float* out      = (float*)d_out;
  float* logp     = out;                                   // [B,V]
  float* o_hidden = out + (size_t)B_ * V_;                 // [1,B,H]
  float* o_ctx2   = o_hidden + B_ * H_;                    // [B,H]
  float* o_aw2    = o_ctx2 + B_ * H_;                      // [B,S]
  float* o_cov2   = o_aw2 + B_ * S_;                       // [B,S]

  float* ws = (float*)d_ws;
  size_t off = 0;
  float* enc_feat = ws + off; off += (size_t)B_ * S_ * H_; // 16.8M fl
  float* dec_feat = ws + off; off += (size_t)B_ * H_;
  float* cov_feat = ws + off; off += (size_t)B_ * S_;
  float* w_raw    = ws + off; off += (size_t)B_ * S_;
  float* w_attn   = ws + off; off += (size_t)B_ * S_;
  float* cov_cur  = ws + off; off += (size_t)B_ * S_;
  float* x_cat    = ws + off; off += (size_t)B_ * (E_ + H_);
  float* x_buf    = ws + off; off += (size_t)B_ * H_;
  float* gi       = ws + off; off += (size_t)B_ * 3 * H_;
  float* gh       = ws + off; off += (size_t)B_ * 3 * H_;
  float* h_new    = ws + off; off += (size_t)B_ * H_;
  float* cat2     = ws + off; off += (size_t)B_ * 2 * H_;
  float* o_pre    = ws + off; off += (size_t)B_ * H_;

  dim3 blk(128);
  auto gemm = [&](const float* A, int lda, const float* Bw, long brs, int bcs,
                  const float* bias, float* C, int ldc, int M, int N, int K) {
    dim3 grid((N + 63) / 64, (M + 63) / 64);
    if (bcs == 1)
      k_gemm_bf16wmma<true><<<grid, blk, 0, stream>>>(A, lda, Bw, brs, bcs, bias, C, ldc, M, N, K);
    else
      k_gemm_bf16wmma<false><<<grid, blk, 0, stream>>>(A, lda, Bw, brs, bcs, bias, C, ldc, M, N, K);
  };

  // enc_feat computed ONCE; shared by both attention calls (halves dominant GEMM)
  gemm(enc, H_, attnLin_w, H_, 1, attnLin_b, enc_feat, H_, B_ * S_, H_, H_);
  k_embed<<<B_, E_, 0, stream>>>(token, emb, x_cat);

  // ---- attention call 1 (hid = hidden[0], cov = coverage)
  gemm(hidden, H_, decAttn_w, H_, 1, decAttn_b, dec_feat, H_, B_, H_, H_);
  gemm(coverage, S_, cvg_w + (H_ - 1) / 2, (long)S_ * H_, H_, cvg_b, cov_feat, S_, B_, S_, S_);
  k_attn_score<<<B_, 256, 0, stream>>>(enc_feat, dec_feat, cov_feat, vvec, mask, w_raw);
  k_softmax_cov<<<B_, 256, 0, stream>>>(w_raw, coverage, w_attn, cov_cur, nullptr);
  k_ctx<<<B_, 256, 0, stream>>>(w_attn, enc, x_cat + E_, E_ + H_, nullptr);

  // x = concat(embedded, ctx1) @ newIn_w^T + b
  gemm(x_cat, E_ + H_, newIn_w, E_ + H_, 1, newIn_b, x_buf, H_, B_, H_, E_ + H_);
  // GRU gates + elementwise step
  gemm(x_buf, H_, gru_w_ih, H_, 1, gru_b_ih, gi, 3 * H_, B_, 3 * H_, H_);
  gemm(hidden, H_, gru_w_hh, H_, 1, gru_b_hh, gh, 3 * H_, B_, 3 * H_, H_);
  k_gru<<<(B_ * H_ + 255) / 256, 256, 0, stream>>>(gi, gh, hidden, h_new, o_hidden, cat2);

  // ---- attention call 2 (hid = h_new, cov = cov_cur) -> aw2/cov2/ctx2 outputs
  gemm(h_new, H_, decAttn_w, H_, 1, decAttn_b, dec_feat, H_, B_, H_, H_);
  gemm(cov_cur, S_, cvg_w + (H_ - 1) / 2, (long)S_ * H_, H_, cvg_b, cov_feat, S_, B_, S_, S_);
  k_attn_score<<<B_, 256, 0, stream>>>(enc_feat, dec_feat, cov_feat, vvec, mask, w_raw);
  k_softmax_cov<<<B_, 256, 0, stream>>>(w_raw, cov_cur, w_attn, o_cov2, o_aw2);
  k_ctx<<<B_, 256, 0, stream>>>(w_attn, enc, cat2 + H_, 2 * H_, o_ctx2);

  // o = concat(h_new, ctx2) @ preOut^T; logits = o @ out_w^T + out_b; log_softmax
  gemm(cat2, 2 * H_, preOut_w, 2 * H_, 1, preOut_b, o_pre, H_, B_, H_, 2 * H_);
  gemm(o_pre, H_, out_w, H_, 1, out_b, logp, V_, B_, V_, H_);
  k_logsoftmax<<<B_, 256, 0, stream>>>(logp);
}
